// dnc_model_6141803233731
// MI455X (gfx1250) — matrix operand order repaired and checked by reading, not compiled, not run on hardware
//
#include <hip/hip_runtime.h>
#include <hip/hip_bf16.h>
#include <math.h>

typedef __attribute__((ext_vector_type(16))) _Float16 v16h;
typedef __attribute__((ext_vector_type(8)))  float    v8f;

#define T_    64
#define B_    32
#define IN_   64
#define OUT_  64
#define N_    128
#define WD_   64
#define R_    4
#define H_    256
#define GN_   1024
#define IFP_  480     // padded interface width (471 -> 480)
#define IFS_  471
#define CLIP_ 20.0f
#define EPS_  1e-6f

__device__ __forceinline__ float sigf(float x){ return 1.0f/(1.0f+expf(-x)); }
__device__ __forceinline__ float softplusf(float x){ return (x > 20.f) ? x : log1pf(expf(x)); }

// ---------------------------------------------------------------------------
// Pack a (possibly vertically-concatenated) f32 weight matrix [r0+r1, ncols]
// into f16 WMMA B-fragments: layout [nt][kt][lane][16 halves].
// B (32x16) lane layout: lane L -> n = L&15, K = kt*32 + (L<16?0:16) + e.
// ---------------------------------------------------------------------------
__global__ void pack_b(const float* __restrict__ s0, int r0,
                       const float* __restrict__ s1, int r1,
                       int ncols, _Float16* __restrict__ dst, int Kt, int Nt)
{
    int idx = blockIdx.x * blockDim.x + threadIdx.x;
    int total = Nt * Kt * 32;
    if (idx >= total) return;
    int lane = idx & 31;
    int kt   = (idx >> 5) % Kt;
    int nt   = idx / (Kt * 32);
    int n     = nt * 16 + (lane & 15);
    int kbase = kt * 32 + ((lane < 16) ? 0 : 16);
    _Float16* out = dst + (size_t)idx * 16;
    #pragma unroll
    for (int e = 0; e < 16; ++e) {
        int k = kbase + e;
        float v = 0.f;
        if (n < ncols) {
            if (k < r0)            v = s0[(size_t)k * ncols + n];
            else if (k - r0 < r1)  v = s1[(size_t)(k - r0) * ncols + n];
        }
        out[e] = (_Float16)v;
    }
}

// ---------------------------------------------------------------------------
// Pack activation rows (up to 3 concatenated f32 sources, each [32, cX]) into
// f16 WMMA A-fragments: layout [mt][kt][lane][16 halves].
// A (16x32) lane layout: lane L -> m = L&15, kbase = (L<16?0:16),
//                        K = kt*32 + kbase + e.
// ---------------------------------------------------------------------------
__global__ void pack_a(const float* __restrict__ s0, int c0,
                       const float* __restrict__ s1, int c1,
                       const float* __restrict__ s2, int c2,
                       _Float16* __restrict__ dst, int Kt)
{
    int idx = blockIdx.x * blockDim.x + threadIdx.x;
    int total = 2 * Kt * 32;
    if (idx >= total) return;
    int lane = idx & 31;
    int kt   = (idx >> 5) % Kt;
    int mt   = idx / (Kt * 32);
    int m  = mt * 16 + (lane & 15);
    int kb = kt * 32 + ((lane < 16) ? 0 : 16);
    _Float16* out = dst + (size_t)idx * 16;
    #pragma unroll
    for (int e = 0; e < 16; ++e) {
        int k = kb + e;
        float v = 0.f;
        if (k < c0)                 v = s0[(size_t)m * c0 + k];
        else if (k - c0 < c1)       v = s1[(size_t)m * c1 + (k - c0)];
        else if (k - c0 - c1 < c2)  v = s2[(size_t)m * c2 + (k - c0 - c1)];
        out[e] = (_Float16)v;
    }
}

// ---------------------------------------------------------------------------
// WMMA GEMM: C[32, Nt*16] = A[32, Kt*32] * B[Kt*32, Nt*16] + bias, clipped.
// One 16x16 tile per wave; fragments are pre-swizzled so each lane issues
// two global_load_b128 per fragment.
// ---------------------------------------------------------------------------
__global__ void gemm_wmma(const _Float16* __restrict__ Af,
                          const _Float16* __restrict__ Bf,
                          const float* __restrict__ bias,
                          float* __restrict__ C,
                          int Kt, int Nt, int ldc, int nact, float clip)
{
    int wid = (blockIdx.x * blockDim.x + threadIdx.x) >> 5;
    if (wid >= 2 * Nt) return;          // wave-uniform exit (EXEC stays all-1)
    int lane = threadIdx.x & 31;
    int nt = wid % Nt;
    int mt = wid / Nt;

    v8f acc = {0.f, 0.f, 0.f, 0.f, 0.f, 0.f, 0.f, 0.f};
    const v16h* ap = (const v16h*)(Af + (size_t)mt * Kt * 512);
    const v16h* bp = (const v16h*)(Bf + (size_t)nt * Kt * 512);
    for (int kt = 0; kt < Kt; ++kt) {
        v16h a = ap[kt * 32 + lane];
        v16h b = bp[kt * 32 + lane];
        acc = __builtin_amdgcn_wmma_f32_16x16x32_f16(
            /*neg_a=*/false, a, /*neg_b=*/false, b,
            /*c_mod=*/(short)0, acc, /*reuse_a=*/false, /*reuse_b=*/false);
    }
    int n    = nt * 16 + (lane & 15);
    int mrow = mt * 16 + ((lane < 16) ? 0 : 8);
    float bv = (n < nact) ? bias[n] : 0.f;
    #pragma unroll
    for (int v = 0; v < 8; ++v) {
        float x = acc[v] + bv;
        x = fminf(fmaxf(x, -clip), clip);
        C[(size_t)(mrow + v) * ldc + n] = x;
    }
}

// ---------------------------------------------------------------------------
// LSTM gate nonlinearity: g[32,1024] -> h,c [32,256]; optional clipped ctrl.
// ---------------------------------------------------------------------------
__global__ void lstm_gate(const float* __restrict__ g, float* __restrict__ h,
                          float* __restrict__ c, float* __restrict__ ctrl)
{
    int idx = blockIdx.x * blockDim.x + threadIdx.x;
    if (idx >= B_ * H_) return;
    int b = idx / H_, j = idx % H_;
    const float* gr = g + (size_t)b * GN_;
    float gi = gr[j], gf = gr[H_ + j], gg = gr[2 * H_ + j], go = gr[3 * H_ + j];
    float c2 = sigf(gf) * c[idx] + sigf(gi) * tanhf(gg);
    float h2 = sigf(go) * tanhf(c2);
    c[idx] = c2;
    h[idx] = h2;
    if (ctrl) ctrl[idx] = fminf(fmaxf(h2, -CLIP_), CLIP_);
}

// ---------------------------------------------------------------------------
// block reductions (128 threads)
// ---------------------------------------------------------------------------
__device__ __forceinline__ float bsum(float v, volatile float* red, int t) {
    red[t] = v; __syncthreads();
    for (int s = 64; s > 0; s >>= 1) { if (t < s) red[t] += red[t + s]; __syncthreads(); }
    float r = red[0]; __syncthreads(); return r;
}
__device__ __forceinline__ float bmaxr(float v, volatile float* red, int t) {
    red[t] = v; __syncthreads();
    for (int s = 64; s > 0; s >>= 1) { if (t < s) red[t] = fmaxf(red[t], red[t + s]); __syncthreads(); }
    float r = red[0]; __syncthreads(); return r;
}

// ---------------------------------------------------------------------------
// DNC memory machinery — one workgroup (128 threads) per batch element.
// Thread t owns memory slot n = t. M (32KB) and link (64KB) staged in LDS.
// ---------------------------------------------------------------------------
__global__ void __launch_bounds__(128)
dnc_memory(const float* __restrict__ itf, float* __restrict__ Mg,
           float* __restrict__ rwg, float* __restrict__ wwg,
           float* __restrict__ linkg, float* __restrict__ precg,
           float* __restrict__ usageg, float* __restrict__ readsg)
{
    __shared__ float Ms[N_][WD_];
    __shared__ float links[N_][N_];
    __shared__ float usage_s[N_], ww_s[N_], prec_s[N_], a_s[N_], wc_s[N_];
    __shared__ float rc_s[R_][N_], rwp[R_][N_], rwn[R_][N_];
    __shared__ float rkey[R_][WD_], wkey[WD_], ers[WD_], wvec[WD_];
    __shared__ float red[N_];
    __shared__ float skey[N_];
    __shared__ int   sidx[N_];
    __shared__ float rstr[R_], freeg[R_], mb[R_], mf[R_], mc[R_];
    __shared__ float wstr, allocg, writeg;

    int b = blockIdx.x, t = threadIdx.x;
    const float* p = itf + (size_t)b * IFP_;

    // ---- load state + interface ----
    for (int w = 0; w < WD_; ++w) Ms[t][w] = Mg[((size_t)b * N_ + t) * WD_ + w];
    for (int j = 0; j < N_; ++j) links[t][j] = linkg[((size_t)b * N_ + t) * N_ + j];
    usage_s[t] = usageg[b * N_ + t];
    ww_s[t]    = wwg[b * N_ + t];
    prec_s[t]  = precg[b * N_ + t];
    for (int r = 0; r < R_; ++r) rwp[r][t] = rwg[((size_t)b * R_ + r) * N_ + t];
    for (int i = t; i < R_ * WD_; i += N_) rkey[i / WD_][i % WD_] = p[i];
    if (t < WD_) { wkey[t] = p[260 + t]; ers[t] = sigf(p[325 + t]); wvec[t] = p[389 + t]; }
    if (t < R_) {
        rstr[t]  = 1.f + softplusf(p[256 + t]);
        freeg[t] = sigf(p[453 + t]);
        float m0 = p[459 + 3 * t], m1 = p[460 + 3 * t], m2 = p[461 + 3 * t];
        float mx = fmaxf(m0, fmaxf(m1, m2));
        float e0 = expf(m0 - mx), e1 = expf(m1 - mx), e2 = expf(m2 - mx);
        float s = e0 + e1 + e2;
        mb[t] = e0 / s; mf[t] = e1 / s; mc[t] = e2 / s;
    }
    if (t == 0) { wstr = 1.f + softplusf(p[324]); allocg = sigf(p[457]); writeg = sigf(p[458]); }
    __syncthreads();

    // ---- usage update (NW == 1) ----
    float u = usage_s[t] + (1.f - usage_s[t]) * ww_s[t];
    float psi = 1.f;
    for (int r = 0; r < R_; ++r) psi *= (1.f - freeg[r] * rwp[r][t]);
    float un = u * psi;
    usage_s[t] = un;
    skey[t] = EPS_ + (1.f - EPS_) * un;
    sidx[t] = t;
    __syncthreads();

    // ---- allocation: bitonic argsort ascending + exclusive cumprod ----
    for (int k = 2; k <= N_; k <<= 1)
        for (int j = k >> 1; j > 0; j >>= 1) {
            int ixj = t ^ j;
            if (ixj > t) {
                bool up = ((t & k) == 0);
                float a0 = skey[t], a1 = skey[ixj];
                if ((a0 > a1) == up) {
                    skey[t] = a1; skey[ixj] = a0;
                    int i0 = sidx[t]; sidx[t] = sidx[ixj]; sidx[ixj] = i0;
                }
            }
            __syncthreads();
        }
    if (t == 0) {
        float pr = 1.f;
        for (int n = 0; n < N_; ++n) {
            float su = skey[n];
            a_s[sidx[n]] = (1.f - su) * pr;
            pr *= su;
        }
    }
    __syncthreads();

    // ---- write content weights (cosine + softmax) ----
    float mn = 0.f, dt = 0.f;
    for (int w = 0; w < WD_; ++w) { float mv = Ms[t][w]; mn += mv * mv; dt += wkey[w] * mv; }
    mn = sqrtf(mn);
    float kn  = sqrtf(bsum((t < WD_) ? wkey[t] * wkey[t] : 0.f, red, t));
    float sim = dt / (kn * mn + EPS_) * wstr;
    float smx = bmaxr(sim, red, t);
    float ex  = expf(sim - smx);
    float ssm = bsum(ex, red, t);
    wc_s[t] = ex / ssm;
    __syncthreads();

    // ---- write weights ----
    float wwn = writeg * (allocg * a_s[t] + (1.f - allocg) * wc_s[t]);
    ww_s[t] = wwn;
    __syncthreads();

    // ---- memory erase/add ----
    for (int w = 0; w < WD_; ++w)
        Ms[t][w] = Ms[t][w] * (1.f - wwn * ers[w]) + wwn * wvec[w];

    // ---- link + precedence update (bsum barrier also fences Ms writes) ----
    float sumww = bsum(wwn, red, t);
    for (int j = 0; j < N_; ++j) {
        float lij = (1.f - wwn - ww_s[j]) * links[t][j] + wwn * prec_s[j];
        links[t][j] = (j == t) ? 0.f : lij;
    }
    __syncthreads();
    float pn = (1.f - sumww) * prec_s[t] + wwn;
    prec_s[t] = pn;

    // ---- read content weights on updated M ----
    float mn2 = 0.f;
    for (int w = 0; w < WD_; ++w) { float mv = Ms[t][w]; mn2 += mv * mv; }
    mn2 = sqrtf(mn2);
    for (int r = 0; r < R_; ++r) {
        float kk = bsum((t < WD_) ? rkey[r][t] * rkey[r][t] : 0.f, red, t);
        float d2 = 0.f;
        for (int w = 0; w < WD_; ++w) d2 += rkey[r][w] * Ms[t][w];
        float s2 = d2 / (sqrtf(kk) * mn2 + EPS_) * rstr[r];
        float m2 = bmaxr(s2, red, t);
        float e2 = expf(s2 - m2);
        float z2 = bsum(e2, red, t);
        rc_s[r][t] = e2 / z2;
    }
    __syncthreads();

    // ---- forward / backward link reads + read-weight mix ----
    float fw[R_] = {0.f, 0.f, 0.f, 0.f}, bw[R_] = {0.f, 0.f, 0.f, 0.f};
    for (int j = 0; j < N_; ++j) {
        float lij = links[t][j], lji = links[j][t];
        #pragma unroll
        for (int r = 0; r < R_; ++r) { fw[r] += lij * rwp[r][j]; bw[r] += lji * rwp[r][j]; }
    }
    for (int r = 0; r < R_; ++r)
        rwn[r][t] = mb[r] * bw[r] + mc[r] * rc_s[r][t] + mf[r] * fw[r];
    __syncthreads();

    // ---- reads = rw_new @ M ----
    for (int o = t; o < R_ * WD_; o += N_) {
        int r = o >> 6, w = o & 63;
        float s = 0.f;
        for (int n = 0; n < N_; ++n) s += rwn[r][n] * Ms[n][w];
        readsg[(size_t)b * R_ * WD_ + o] = s;
    }

    // ---- write back state ----
    for (int w = 0; w < WD_; ++w) Mg[((size_t)b * N_ + t) * WD_ + w] = Ms[t][w];
    for (int j = 0; j < N_; ++j) linkg[((size_t)b * N_ + t) * N_ + j] = links[t][j];
    usageg[b * N_ + t] = usage_s[t];
    wwg[b * N_ + t]    = ww_s[t];
    precg[b * N_ + t]  = prec_s[t];
    for (int r = 0; r < R_; ++r) rwg[((size_t)b * R_ + r) * N_ + t] = rwn[r][t];
}

// ---------------------------------------------------------------------------
extern "C" void kernel_launch(void* const* d_in, const int* in_sizes, int n_in,
                              void* d_out, int out_size, void* d_ws, size_t ws_size,
                              hipStream_t stream)
{
    (void)in_sizes; (void)n_in; (void)out_size; (void)ws_size;
    const float* inputs = (const float*)d_in[0];
    const float* Wx0  = (const float*)d_in[1];
    const float* Wh0  = (const float*)d_in[2];
    const float* b0   = (const float*)d_in[3];
    const float* Wx1  = (const float*)d_in[4];
    const float* Wh1  = (const float*)d_in[5];
    const float* b1   = (const float*)d_in[6];
    const float* Wif  = (const float*)d_in[7];
    const float* bif  = (const float*)d_in[8];
    const float* Wout = (const float*)d_in[9];
    const float* bout = (const float*)d_in[10];
    float* out = (float*)d_out;

    char* ws = (char*)d_ws;
    size_t off = 0;
    auto alloc = [&](size_t bytes) -> size_t {
        size_t o = off; off += (bytes + 255) & ~(size_t)255; return o;
    };

    // packed weights (f16 fragments)
    size_t oW0p   = alloc((size_t)64 * 18 * 512 * 2);
    size_t oW1p   = alloc((size_t)64 * 16 * 512 * 2);
    size_t oWifp  = alloc((size_t)30 * 8 * 512 * 2);
    size_t oWoutp = alloc((size_t)4 * 16 * 512 * 2);
    // per-step activation fragments (f16)
    size_t oA0   = alloc((size_t)2 * 18 * 512 * 2);
    size_t oA1   = alloc((size_t)2 * 16 * 512 * 2);
    size_t oAif  = alloc((size_t)2 * 8 * 512 * 2);
    size_t oAout = alloc((size_t)2 * 16 * 512 * 2);
    // f32 intermediates
    size_t oG0  = alloc((size_t)32 * 1024 * 4);
    size_t oG1  = alloc((size_t)32 * 1024 * 4);
    size_t oItf = alloc((size_t)32 * IFP_ * 4);
    // recurrent state (zeroed every launch)
    size_t stateBegin = off;
    size_t oH0   = alloc((size_t)B_ * H_ * 4);
    size_t oC0   = alloc((size_t)B_ * H_ * 4);
    size_t oH1   = alloc((size_t)B_ * H_ * 4);
    size_t oC1   = alloc((size_t)B_ * H_ * 4);
    size_t oCtrl = alloc((size_t)B_ * H_ * 4);
    size_t oM    = alloc((size_t)B_ * N_ * WD_ * 4);
    size_t oRw   = alloc((size_t)B_ * R_ * N_ * 4);
    size_t oWw   = alloc((size_t)B_ * N_ * 4);
    size_t oLink = alloc((size_t)B_ * N_ * N_ * 4);
    size_t oPrec = alloc((size_t)B_ * N_ * 4);
    size_t oUse  = alloc((size_t)B_ * N_ * 4);
    size_t oRds  = alloc((size_t)B_ * R_ * WD_ * 4);
    size_t stateEnd = off;

    _Float16* W0p   = (_Float16*)(ws + oW0p);
    _Float16* W1p   = (_Float16*)(ws + oW1p);
    _Float16* Wifp  = (_Float16*)(ws + oWifp);
    _Float16* Woutp = (_Float16*)(ws + oWoutp);
    _Float16* A0   = (_Float16*)(ws + oA0);
    _Float16* A1   = (_Float16*)(ws + oA1);
    _Float16* Aif  = (_Float16*)(ws + oAif);
    _Float16* Aout = (_Float16*)(ws + oAout);
    float* g0   = (float*)(ws + oG0);
    float* g1   = (float*)(ws + oG1);
    float* itf  = (float*)(ws + oItf);
    float* h0   = (float*)(ws + oH0);
    float* c0   = (float*)(ws + oC0);
    float* h1   = (float*)(ws + oH1);
    float* c1   = (float*)(ws + oC1);
    float* ctrl = (float*)(ws + oCtrl);
    float* Mm   = (float*)(ws + oM);
    float* rw   = (float*)(ws + oRw);
    float* wwv  = (float*)(ws + oWw);
    float* link = (float*)(ws + oLink);
    float* prec = (float*)(ws + oPrec);
    float* use  = (float*)(ws + oUse);
    float* reads = (float*)(ws + oRds);

    // zero recurrent state (deterministic across calls / graph replays)
    hipMemsetAsync(ws + stateBegin, 0, stateEnd - stateBegin, stream);

    // pack weights (concatenated-K GEMMs: W0=[Wx0;Wh0], W1=[Wx1;Wh1])
    pack_b<<<(64 * 18 * 32 + 255) / 256, 256, 0, stream>>>(Wx0, 320, Wh0, 256, 1024, W0p, 18, 64);
    pack_b<<<(64 * 16 * 32 + 255) / 256, 256, 0, stream>>>(Wx1, 256, Wh1, 256, 1024, W1p, 16, 64);
    pack_b<<<(30 * 8 * 32 + 255) / 256, 256, 0, stream>>>(Wif, 256, nullptr, 0, IFS_, Wifp, 8, 30);
    pack_b<<<(4 * 16 * 32 + 255) / 256, 256, 0, stream>>>(Wout, 512, nullptr, 0, 64, Woutp, 16, 4);

    for (int st = 0; st < T_; ++st) {
        const float* xt = inputs + (size_t)st * B_ * IN_;

        // LSTM0: [x_t | reads | h0] (K=576) @ [Wx0;Wh0] -> gates -> h0,c0
        pack_a<<<(2 * 18 * 32 + 255) / 256, 256, 0, stream>>>(xt, IN_, reads, R_ * WD_, h0, H_, A0, 18);
        gemm_wmma<<<32, 128, 0, stream>>>(A0, W0p, b0, g0, 18, 64, 1024, 1024, 1e30f);
        lstm_gate<<<32, 256, 0, stream>>>(g0, h0, c0, nullptr);

        // LSTM1: [h0 | h1] (K=512) @ [Wx1;Wh1] -> gates -> h1,c1, ctrl=clip(h1)
        pack_a<<<4, 256, 0, stream>>>(h0, H_, h1, H_, nullptr, 0, A1, 16);
        gemm_wmma<<<32, 128, 0, stream>>>(A1, W1p, b1, g1, 16, 64, 1024, 1024, 1e30f);
        lstm_gate<<<32, 256, 0, stream>>>(g1, h1, c1, ctrl);

        // interface: ctrl (K=256) @ W_if -> itf [32,480] (padded)
        pack_a<<<2, 256, 0, stream>>>(ctrl, H_, nullptr, 0, nullptr, 0, Aif, 8);
        gemm_wmma<<<15, 128, 0, stream>>>(Aif, Wifp, bif, itf, 8, 30, IFP_, IFS_, 1e30f);

        // DNC memory machinery: one workgroup per batch element
        dnc_memory<<<32, 128, 0, stream>>>(itf, Mm, rw, wwv, link, prec, use, reads);

        // output: [ctrl | reads] (K=512) @ W_out -> clipped out[t]
        pack_a<<<4, 256, 0, stream>>>(ctrl, H_, reads, R_ * WD_, nullptr, 0, Aout, 16);
        gemm_wmma<<<2, 128, 0, stream>>>(Aout, Woutp, bout, out + (size_t)st * B_ * OUT_,
                                         16, 4, OUT_, OUT_, CLIP_);
    }
}
